// MultiHeadSelfAttention_66357244723363
// MI455X (gfx1250) — matrix operand order repaired and checked by reading, not compiled, not run on hardware
//
#include <hip/hip_runtime.h>
#include <hip/hip_bf16.h>
#include <math.h>
#include <cstdint>

// ---------------------------------------------------------------------------
// MHSA forward for MI455X (gfx1250): bf16 WMMA everywhere, flash softmax.
// B=4, T=2048, C=1024, H=16, D=64. Compute-bound (~137 GFLOP vs ~50MB I/O).
// GEMMs: LDS-staged, double-buffered B tiles via async-to-LDS copies.
// ---------------------------------------------------------------------------

typedef __bf16 bf16_t;
typedef __attribute__((ext_vector_type(16))) __bf16 v16bf;
typedef __attribute__((ext_vector_type(8)))  __bf16 v8bf;
typedef __attribute__((ext_vector_type(8)))  float  v8f;
typedef __attribute__((ext_vector_type(4)))  int    v4i;

constexpr int Bc = 4, Tc = 2048, Cc = 1024, Hc = 16, Dc = 64;
constexpr int Mrows = Bc * Tc;        // 8192 token rows

// ---- CDNA5 async global->LDS copy (ASYNCcnt), guarded by __has_builtin ----
#if __has_builtin(__builtin_amdgcn_global_load_async_to_lds_b128)
#define HAVE_ASYNC_COPY 1
#else
#define HAVE_ASYNC_COPY 0
#endif

typedef __attribute__((address_space(1))) v4i* gv4i_p;
typedef __attribute__((address_space(3))) v4i* lv4i_p;

__device__ __forceinline__ void copy16_to_lds(const bf16_t* g, bf16_t* l) {
#if HAVE_ASYNC_COPY
  __builtin_amdgcn_global_load_async_to_lds_b128(
      (gv4i_p)(uintptr_t)g,
      (lv4i_p)(uint32_t)(uintptr_t)l,
      0, 0);
#else
  *(v8bf*)l = *(const v8bf*)g;       // synchronous fallback
#endif
}

__device__ __forceinline__ void async_wait0() {
#if HAVE_ASYNC_COPY
#if __has_builtin(__builtin_amdgcn_s_wait_asynccnt)
  __builtin_amdgcn_s_wait_asynccnt(0);
#else
  asm volatile("s_wait_asynccnt 0" ::: "memory");
#endif
#endif
}

// Concatenate two 8-wide bf16 vectors into the 16-wide WMMA operand.
__device__ __forceinline__ v16bf cat8(v8bf lo, v8bf hi) {
  return __builtin_shufflevector(lo, hi, 0,1,2,3,4,5,6,7,8,9,10,11,12,13,14,15);
}

// A fragment: 16x32 bf16 tile, row-major source with leading dim ld.
// ISA layout: lanes 0-15 hold row M=lane with K in {0..7,16..23};
//             lanes 16-31 hold row M=lane-16 with K in {8..15,24..31}.
__device__ __forceinline__ v16bf load_a_frag(const bf16_t* base, int ld) {
  const int lane = threadIdx.x & 31;
  const int half = lane >> 4;
  const int m    = lane & 15;
  const bf16_t* row = base + (size_t)m * ld;
  const int k0 = half ? 8 : 0;                 // first 8 K's
  v8bf lo = *(const v8bf*)(row + k0);          // K = k0 .. k0+7
  v8bf hi = *(const v8bf*)(row + k0 + 16);     // K = k0+16 .. k0+23
  return cat8(lo, hi);
}

// B fragment: 32x16 bf16 tile where B[k][n] = W[n][k], W row-major [N][K].
// ISA layout: lane n (0-15) holds column N=n, K in {0..7,16..23}; lane n+16 holds K in {8..15,24..31}.
__device__ __forceinline__ v16bf load_b_frag(const bf16_t* base, int ld) {
  const int lane = threadIdx.x & 31;
  const int half = lane >> 4;
  const int nn   = lane & 15;
  const bf16_t* row = base + (size_t)nn * ld + half * 8;
  v8bf lo = *(const v8bf*)(row);
  v8bf hi = *(const v8bf*)(row + 16);
  return cat8(lo, hi);
}

#define WMMA_BF16(a, b, c) \
  __builtin_amdgcn_wmma_f32_16x16x32_bf16(false, (a), false, (b), (short)0, (c), false, false)

// ---------------------------------------------------------------------------
// fp32 -> bf16 conversion (grid-stride)
// ---------------------------------------------------------------------------
__global__ void cvt_bf16_kernel(const float* __restrict__ src,
                                bf16_t* __restrict__ dst, int nelem) {
  int i = blockIdx.x * blockDim.x + threadIdx.x;
  int stride = gridDim.x * blockDim.x;
  for (; i < nelem; i += stride) dst[i] = (bf16_t)src[i];
}

// ---------------------------------------------------------------------------
// Cooperative 64x32 bf16 B-tile fill: 256 threads x 16B = 4KB.
// ---------------------------------------------------------------------------
__device__ __forceinline__ void fill_b_tile(const bf16_t* W, int col0, int kk,
                                            bf16_t* dst, int ldw) {
  const int t   = threadIdx.x;     // 0..255
  const int n   = t >> 2;          // 0..63  (output column)
  const int seg = (t & 3) * 8;     // 0/8/16/24 (k segment)
  copy16_to_lds(W + (size_t)(col0 + n) * ldw + kk + seg, dst + n * 32 + seg);
}

// Shared GEMM core: 8 waves, block tile 256(rows) x 64(cols), K = Cc.
// Each wave: 32x64 via 2 A-frags x 4 shared B-frags = 8 WMMAs per k-step.
// B tiles double-buffered in LDS (async copies); A register-prefetched.
__device__ __forceinline__ void gemm_core(const bf16_t* __restrict__ Ab,
                                          const bf16_t* __restrict__ Wb,
                                          bf16_t* sB /* [2][64*32] */,
                                          int row0, int col0, v8f acc[2][4]) {
  fill_b_tile(Wb, col0, 0, sB, Cc);
  v16bf a0 = load_a_frag(Ab + (size_t)row0 * Cc, Cc);
  v16bf a1 = load_a_frag(Ab + (size_t)(row0 + 16) * Cc, Cc);
  async_wait0();
  __syncthreads();

  int cur = 0;
  for (int kk = 0; kk < Cc; kk += 32) {
    v16bf aN0 = a0, aN1 = a1;
    const bool more = (kk + 32) < Cc;
    if (more) {   // prefetch next B tile (async->LDS) and next A frags (regs)
      fill_b_tile(Wb, col0, kk + 32, sB + (cur ^ 1) * (64 * 32), Cc);
      aN0 = load_a_frag(Ab + (size_t)row0 * Cc + kk + 32, Cc);
      aN1 = load_a_frag(Ab + (size_t)(row0 + 16) * Cc + kk + 32, Cc);
    }
    const bf16_t* bt = sB + cur * (64 * 32);
#pragma unroll
    for (int j = 0; j < 4; ++j) {
      v16bf b = load_b_frag(bt + j * 16 * 32, 32);
      acc[0][j] = WMMA_BF16(a0, b, acc[0][j]);
      acc[1][j] = WMMA_BF16(a1, b, acc[1][j]);
    }
    async_wait0();
    __syncthreads();
    cur ^= 1;
    a0 = aN0; a1 = aN1;
  }
}

// ---------------------------------------------------------------------------
// QKV projection: qkv[m,f] = x[m,:] . qkv_w[f,:] + qkv_b[f]
// Scatter into Q[B,H,T,D], K[B,H,T,D], and V transposed [B,H,D,T].
// ---------------------------------------------------------------------------
__global__ __launch_bounds__(256)
void qkv_gemm_kernel(const bf16_t* __restrict__ Xb, const bf16_t* __restrict__ Wb,
                     const float* __restrict__ bias,
                     bf16_t* __restrict__ Qb, bf16_t* __restrict__ Kb,
                     bf16_t* __restrict__ VTb) {
  __shared__ __align__(16) bf16_t sB[2 * 64 * 32];
  const int wave = threadIdx.x >> 5;
  const int lane = threadIdx.x & 31;
  const int half = lane >> 4;
  const int nidx = lane & 15;
  const int row0 = blockIdx.x * 256 + wave * 32;
  const int col0 = blockIdx.y * 64;

  v8f acc[2][4] = {};
  gemm_core(Xb, Wb, sB, row0, col0, acc);

#pragma unroll
  for (int i = 0; i < 2; ++i) {
#pragma unroll
    for (int j = 0; j < 4; ++j) {
#pragma unroll
      for (int r = 0; r < 8; ++r) {
        int mg = row0 + i * 16 + r + half * 8;   // global token row
        int f  = col0 + j * 16 + nidx;           // output feature (0..3071)
        float v = acc[i][j][r] + bias[f];
        int which = f >> 10;                     // 0=Q 1=K 2=V
        int rem   = f & (Cc - 1);
        int h = rem >> 6, d = rem & (Dc - 1);
        int bb = mg >> 11;                       // /T
        int t  = mg & (Tc - 1);
        size_t bh = (size_t)(bb * Hc + h);
        bf16_t val = (bf16_t)v;
        if (which == 0)      Qb[(bh * Tc + t) * Dc + d] = val;
        else if (which == 1) Kb[(bh * Tc + t) * Dc + d] = val;
        else                 VTb[(bh * Dc + d) * Tc + t] = val;  // V^T
      }
    }
  }
}

// ---------------------------------------------------------------------------
// Flash attention: one wave owns a 16-row query tile; stream 32-key chunks.
// All 8 K/V fragments of a chunk load in one clause; online softmax; P staged
// through wave-private LDS (C-layout -> A-layout) guarded by s_wait_dscnt.
// ---------------------------------------------------------------------------
__global__ __launch_bounds__(128)
void flash_attn_kernel(const bf16_t* __restrict__ Qb, const bf16_t* __restrict__ Kb,
                       const bf16_t* __restrict__ VTb, bf16_t* __restrict__ Ob) {
  __shared__ __align__(16) bf16_t plds[4 * 16 * 32];   // per-wave P staging
  const int wave = threadIdx.x >> 5;
  const int lane = threadIdx.x & 31;
  const int half = lane >> 4;
  const int nidx = lane & 15;
  const int b = blockIdx.z, h = blockIdx.y;
  const int qbase = blockIdx.x * 64 + wave * 16;
  const size_t bh = (size_t)(b * Hc + h);
  const bf16_t* Qp = Qb + (bh * Tc + qbase) * Dc;
  const bf16_t* Kp = Kb + bh * Tc * Dc;
  const bf16_t* Vp = VTb + bh * Dc * Tc;               // [D][T]
  bf16_t* myp = plds + wave * (16 * 32);

  v16bf a0 = load_a_frag(Qp, Dc);
  v16bf a1 = load_a_frag(Qp + 32, Dc);

  v8f O[4] = {};
  float mrow[8], lrow[8];
#pragma unroll
  for (int r = 0; r < 8; ++r) { mrow[r] = -1e30f; lrow[r] = 0.f; }
  const float scale = 0.125f;   // 1/sqrt(64)

  for (int k0 = 0; k0 < qbase + 16; k0 += 32) {
    // hoist the whole chunk's K and V fragments into one load clause
    v16bf kb00 = load_b_frag(Kp + (size_t)k0 * Dc, Dc);
    v16bf kb01 = load_b_frag(Kp + (size_t)k0 * Dc + 32, Dc);
    v16bf kb10 = load_b_frag(Kp + (size_t)(k0 + 16) * Dc, Dc);
    v16bf kb11 = load_b_frag(Kp + (size_t)(k0 + 16) * Dc + 32, Dc);
    v16bf vb[4];
#pragma unroll
    for (int j = 0; j < 4; ++j)
      vb[j] = load_b_frag(Vp + (size_t)(j * 16) * Tc + k0, Tc);

    v8f s[2];
    {
      v8f c0 = {}, c1 = {};
      c0 = WMMA_BF16(a0, kb00, c0);
      c1 = WMMA_BF16(a0, kb10, c1);
      c0 = WMMA_BF16(a1, kb01, c0);
      c1 = WMMA_BF16(a1, kb11, c1);
      s[0] = c0; s[1] = c1;
    }
    // scale + causal mask (row = qbase + r + half*8, col = k0 + t2*16 + n)
#pragma unroll
    for (int t2 = 0; t2 < 2; ++t2) {
      int colg = k0 + t2 * 16 + nidx;
#pragma unroll
      for (int r = 0; r < 8; ++r) {
        int rowg = qbase + r + half * 8;
        float v = s[t2][r] * scale;
        s[t2][r] = (colg <= rowg) ? v : -1e30f;
      }
    }
    // online softmax (row reductions across the 16 lanes of each half)
    float alpha[8];
#pragma unroll
    for (int r = 0; r < 8; ++r) {
      float tm = fmaxf(s[0][r], s[1][r]);
#pragma unroll
      for (int msk = 1; msk < 16; msk <<= 1) tm = fmaxf(tm, __shfl_xor(tm, msk, 32));
      float mnew = fmaxf(mrow[r], tm);
      alpha[r] = __expf(mrow[r] - mnew);
      float p0 = __expf(s[0][r] - mnew);
      float p1 = __expf(s[1][r] - mnew);
      s[0][r] = p0; s[1][r] = p1;
      float rs = p0 + p1;
#pragma unroll
      for (int msk = 1; msk < 16; msk <<= 1) rs += __shfl_xor(rs, msk, 32);
      lrow[r] = alpha[r] * lrow[r] + rs;
      mrow[r] = mnew;
    }
    // stage P (16x32) via wave-private LDS: C-layout -> A-layout transpose
#pragma unroll
    for (int r = 0; r < 8; ++r) {
      int rl = r + half * 8;
      myp[rl * 32 + nidx]      = (bf16_t)s[0][r];
      myp[rl * 32 + nidx + 16] = (bf16_t)s[1][r];
    }
    asm volatile("s_wait_dscnt 0" ::: "memory");
    v16bf pa = load_a_frag(myp, 32);
    // O = diag(alpha)*O + P @ V
#pragma unroll
    for (int j = 0; j < 4; ++j) {
#pragma unroll
      for (int r = 0; r < 8; ++r) O[j][r] *= alpha[r];
      O[j] = WMMA_BF16(pa, vb[j], O[j]);
    }
  }
  // normalize and emit attn output as bf16 in [B,T,C] (heads re-interleaved)
#pragma unroll
  for (int j = 0; j < 4; ++j) {
#pragma unroll
    for (int r = 0; r < 8; ++r) {
      int rl = r + half * 8;
      int t  = qbase + rl;
      int cc = h * Dc + j * 16 + nidx;
      float v = O[j][r] / lrow[r];
      Ob[((size_t)b * Tc + t) * Cc + cc] = (bf16_t)v;
    }
  }
}

// ---------------------------------------------------------------------------
// Output projection: out[m,f] = attn[m,:] . out_w[f,:] + out_b[f]  (fp32 out)
// ---------------------------------------------------------------------------
__global__ __launch_bounds__(256)
void out_gemm_kernel(const bf16_t* __restrict__ Ab, const bf16_t* __restrict__ Wb,
                     const float* __restrict__ bias, float* __restrict__ out) {
  __shared__ __align__(16) bf16_t sB[2 * 64 * 32];
  const int wave = threadIdx.x >> 5;
  const int lane = threadIdx.x & 31;
  const int half = lane >> 4;
  const int nidx = lane & 15;
  const int row0 = blockIdx.x * 256 + wave * 32;
  const int col0 = blockIdx.y * 64;

  v8f acc[2][4] = {};
  gemm_core(Ab, Wb, sB, row0, col0, acc);

#pragma unroll
  for (int i = 0; i < 2; ++i) {
#pragma unroll
    for (int j = 0; j < 4; ++j) {
#pragma unroll
      for (int r = 0; r < 8; ++r) {
        int mg = row0 + i * 16 + r + half * 8;
        int f  = col0 + j * 16 + nidx;
        out[(size_t)mg * Cc + f] = acc[i][j][r] + bias[f];
      }
    }
  }
}

// ---------------------------------------------------------------------------
// Host launcher
// ---------------------------------------------------------------------------
extern "C" void kernel_launch(void* const* d_in, const int* in_sizes, int n_in,
                              void* d_out, int out_size, void* d_ws, size_t ws_size,
                              hipStream_t stream) {
  (void)in_sizes; (void)n_in; (void)out_size; (void)ws_size;
  const float* x     = (const float*)d_in[0];
  const float* qkv_w = (const float*)d_in[1];
  const float* qkv_b = (const float*)d_in[2];
  const float* out_w = (const float*)d_in[3];
  const float* out_b = (const float*)d_in[4];
  float* out = (float*)d_out;

  char* ws = (char*)d_ws;
  size_t off = 0;
  auto carve = [&](size_t bytes) -> char* {
    char* p = ws + off;
    off += (bytes + 255) & ~(size_t)255;
    return p;
  };
  bf16_t* xb   = (bf16_t*)carve((size_t)Mrows * Cc * 2);       // 16 MB
  bf16_t* wqkb = (bf16_t*)carve((size_t)3 * Cc * Cc * 2);      //  6 MB
  bf16_t* wob  = (bf16_t*)carve((size_t)Cc * Cc * 2);          //  2 MB
  bf16_t* Qb   = (bf16_t*)carve((size_t)Mrows * Cc * 2);       // 16 MB
  bf16_t* Kb   = (bf16_t*)carve((size_t)Mrows * Cc * 2);       // 16 MB
  bf16_t* VTb  = (bf16_t*)carve((size_t)Mrows * Cc * 2);       // 16 MB
  bf16_t* Ob   = (bf16_t*)carve((size_t)Mrows * Cc * 2);       // 16 MB

  cvt_bf16_kernel<<<1024, 256, 0, stream>>>(x, xb, Mrows * Cc);
  cvt_bf16_kernel<<<1024, 256, 0, stream>>>(qkv_w, wqkb, 3 * Cc * Cc);
  cvt_bf16_kernel<<<1024, 256, 0, stream>>>(out_w, wob, Cc * Cc);

  qkv_gemm_kernel<<<dim3(Mrows / 256, (3 * Cc) / 64), 256, 0, stream>>>(
      xb, wqkb, qkv_b, Qb, Kb, VTb);

  flash_attn_kernel<<<dim3(Tc / 64, Hc, Bc), 128, 0, stream>>>(Qb, Kb, VTb, Ob);

  out_gemm_kernel<<<dim3(Mrows / 256, Cc / 64), 256, 0, stream>>>(
      Ob, wob, out_b, out);
}
